// GRLSTM_44427141710053
// MI455X (gfx1250) — matrix operand order repaired and checked by reading, not compiled, not run on hardware
//
#include <hip/hip_runtime.h>
#include <hip/hip_bf16.h>
#include <math.h>

#define DEV __device__ __forceinline__

typedef __attribute__((ext_vector_type(16))) _Float16 v16h;
typedef __attribute__((ext_vector_type(8)))  float    v8f;
typedef __attribute__((ext_vector_type(4)))  unsigned v4u;
typedef __attribute__((ext_vector_type(8)))  int      v8i;
typedef __attribute__((ext_vector_type(4)))  int      v4i;

namespace {
constexpr int Bsz = 32, Tseq = 128, Dm = 128;
constexpr int Kn = 10;
constexpr float W_SPA_C = 0.3f / 0.8f;   // 0.375
constexpr float W_SEM_C = 0.5f / 0.8f;   // 0.625
constexpr int MHA_LDS = 294912;          // 288 KB dynamic LDS
}

// ---------------- Tensor Data Mover: 2D f32 tile -> LDS ----------------
// Builds a D# (CDNA5 ISA) and issues TENSOR_LOAD_TO_LDS.
// tile = tw x th f32 elements, row stride `stride` elements, packed into LDS.
DEV void tdm_load_tile_f32(unsigned lds_off, const float* gptr, unsigned tw,
                           unsigned th, unsigned stride) {
  unsigned long long ga = (unsigned long long)(uintptr_t)gptr;
  v4u g0;
  g0[0] = 1u;                                   // count=1 valid descriptor
  g0[1] = lds_off;                              // lds_addr (bytes)
  g0[2] = (unsigned)ga;                         // global_addr[31:0]
  g0[3] = (unsigned)((ga >> 32) & 0x01FFFFFFu)  // global_addr[56:32]
          | 0x80000000u;                        // type=2 ("image")
  v8i g1;
  g1[0] = (int)(2u << 16);                      // data_size=4B, no multicast
  g1[1] = (int)((tw & 0xFFFFu) << 16);          // tensor_dim0[15:0]
  g1[2] = (int)((tw >> 16) | ((th & 0xFFFFu) << 16));   // dim0 hi | dim1 lo
  g1[3] = (int)((th >> 16) | ((tw & 0xFFFFu) << 16));   // dim1 hi | tile_dim0
  g1[4] = (int)(th & 0xFFFFu);                  // tile_dim1 (tile_dim2=0)
  g1[5] = (int)stride;                          // tensor_dim0_stride[31:0]
  g1[6] = 0;                                    // stride hi | dim1_stride lo
  g1[7] = 0;
  v4i z4 = {0, 0, 0, 0};                        // groups 2/3 unused (2D)
  v8i z8 = {0, 0, 0, 0, 0, 0, 0, 0};
  __builtin_amdgcn_tensor_load_to_lds(g0, g1, z4, z4, z8, 0);
}

DEV unsigned lds_offset_of(const void* p) {
  return (unsigned)(uintptr_t)p;  // low 32 bits of flat LDS addr = byte offset
}

// ---------------- WMMA fragment helpers (ISA 7.12.2 layouts) ----------------

DEV v8f wmma_f16(v16h a, v16h b, v8f c) {
  return __builtin_amdgcn_wmma_f32_16x16x32_f16(false, a, false, b, (short)0, c,
                                                false, false);
}

// A 16x32 f16 fragment from f16 src[m][k]
DEV v16h loadA_f16(const _Float16* src, int ld, int row0, int k0) {
  int lane = threadIdx.x & 31;
  int m = row0 + (lane & 15), hf = lane >> 4;
  v16h a;
#pragma unroll
  for (int v = 0; v < 8; ++v)
#pragma unroll
    for (int s = 0; s < 2; ++s) {
      int K = ((v & 4) ? 16 : 0) + hf * 8 + (v & 3) * 2 + s;
      a[v * 2 + s] = src[m * ld + k0 + K];
    }
  return a;
}

// A fragment from f16 src with zero-padding for K >= kmax (dh=16 case)
DEV v16h loadA_f16_pad(const _Float16* src, int ld, int row0, int k0, int kmax) {
  int lane = threadIdx.x & 31;
  int m = row0 + (lane & 15), hf = lane >> 4;
  v16h a;
#pragma unroll
  for (int v = 0; v < 8; ++v)
#pragma unroll
    for (int s = 0; s < 2; ++s) {
      int K = ((v & 4) ? 16 : 0) + hf * 8 + (v & 3) * 2 + s;
      a[v * 2 + s] = (K < kmax) ? src[m * ld + k0 + K] : (_Float16)0.0f;
    }
  return a;
}

// A fragment converting from f32 src[m][k]
DEV v16h loadA_f32cvt(const float* src, int ld, int row0, int k0) {
  int lane = threadIdx.x & 31;
  int m = row0 + (lane & 15), hf = lane >> 4;
  v16h a;
#pragma unroll
  for (int v = 0; v < 8; ++v)
#pragma unroll
    for (int s = 0; s < 2; ++s) {
      int K = ((v & 4) ? 16 : 0) + hf * 8 + (v & 3) * 2 + s;
      a[v * 2 + s] = (_Float16)src[m * ld + k0 + K];
    }
  return a;
}

// B 32x16 fragment, B[k][n] = W[n][k]  (f32 [N,K] row-major, global or LDS)
DEV v16h loadB_wT(const float* W, int ldw, int n0, int k0) {
  int lane = threadIdx.x & 31;
  int n = n0 + (lane & 15), hf = lane >> 4;
  v16h b;
#pragma unroll
  for (int v = 0; v < 8; ++v)
#pragma unroll
    for (int s = 0; s < 2; ++s) {
      int K = ((v & 4) ? 16 : 0) + hf * 8 + (v & 3) * 2 + s;
      b[v * 2 + s] = (_Float16)W[n * ldw + k0 + K];
    }
  return b;
}

// B fragment from k-major f16 matrix M[k][n] (e.g. V: rows = keys)
DEV v16h loadB_kn_f16(const _Float16* Mk, int ld, int n0, int k0) {
  int lane = threadIdx.x & 31;
  int n = n0 + (lane & 15), hf = lane >> 4;
  v16h b;
#pragma unroll
  for (int v = 0; v < 8; ++v)
#pragma unroll
    for (int s = 0; s < 2; ++s) {
      int K = ((v & 4) ? 16 : 0) + hf * 8 + (v & 3) * 2 + s;
      b[v * 2 + s] = Mk[(k0 + K) * ld + n];
    }
  return b;
}

// B fragment for K^T: B[k][n] = Kmat[n][k], zero-pad K >= kmax (dh=16)
DEV v16h loadB_nk_f16_pad(const _Float16* Mn, int ld, int n0, int k0, int kmax) {
  int lane = threadIdx.x & 31;
  int n = n0 + (lane & 15), hf = lane >> 4;
  v16h b;
#pragma unroll
  for (int v = 0; v < 8; ++v)
#pragma unroll
    for (int s = 0; s < 2; ++s) {
      int K = ((v & 4) ? 16 : 0) + hf * 8 + (v & 3) * 2 + s;
      b[v * 2 + s] = (K < kmax) ? Mn[n * ld + k0 + K] : (_Float16)0.0f;
    }
  return b;
}

DEV void storeC_f32(float* dst, int ld, int row0, int col0, v8f c) {
  int lane = threadIdx.x & 31;
  int n = col0 + (lane & 15), hf = lane >> 4;
#pragma unroll
  for (int v = 0; v < 8; ++v) dst[(row0 + v + hf * 8) * ld + n] = c[v];
}

// ---------------- misc device helpers ----------------

DEV float block_sum128(float v, float* red) {  // blockDim.x == 128
  int d = threadIdx.x;
  red[d] = v;
  __syncthreads();
#pragma unroll
  for (int s = 64; s > 0; s >>= 1) {
    if (d < s) red[d] += red[d + s];
    __syncthreads();
  }
  float r = red[0];
  __syncthreads();
  return r;
}

DEV float gelu_tanh(float x) {
  float x3 = x * x * x;
  return 0.5f * x * (1.0f + tanhf(0.7978845608028654f * (x + 0.044715f * x3)));
}

// ---------------- spatial branch: only the 4096 needed nodes ----------------

__global__ __launch_bounds__(128) void spatial_kernel(
    const float* __restrict__ node_emb, const float* __restrict__ rel_emb,
    const int* __restrict__ nodes, const int* __restrict__ nbr_idx,
    const int* __restrict__ nbr_rel, float* __restrict__ spa_seq) {
  __shared__ float rel[10 * Dm];
  __shared__ int ids[11];
  __shared__ int rels[11 * Kn];
  __shared__ float red[Dm];
  int bt = blockIdx.x;
  int d = threadIdx.x;
  for (int i = d; i < 10 * Dm; i += Dm) rel[i] = rel_emb[i];
  if (d == 0) {
    int n = nodes[bt];
    ids[0] = n;
#pragma unroll
    for (int k = 0; k < Kn; ++k) ids[1 + k] = nbr_idx[n * Kn + k];
  }
  __syncthreads();
  if (d < 11) {
    int m = ids[d];
#pragma unroll
    for (int k = 0; k < Kn; ++k) rels[d * Kn + k] = nbr_rel[m * Kn + k];
  }
  __syncthreads();

  float nf[11];
#pragma unroll
  for (int j = 0; j < 11; ++j) {
    float rb = 0.0f;
#pragma unroll
    for (int k = 0; k < Kn; ++k) rb += rel[rels[j * Kn + k] * Dm + d];
    nf[j] = node_emb[ids[j] * Dm + d] + rb * 0.1f;
  }
  float acc = nf[0] + rel[9 * Dm + d];  // self edge (SELF_REL=9)
#pragma unroll
  for (int k = 0; k < Kn; ++k) acc += nf[1 + k] + rel[rels[k] * Dm + d];
  float spa = acc * (1.0f / 11.0f) + nf[0];

  float n2 = block_sum128(spa * spa, red);
  float nrm = fmaxf(sqrtf(n2), 1e-12f);
  spa = spa * (11.313708498984760f / nrm);  // * sqrt(128)

  int t = bt & (Tseq - 1);
  float j2 = (float)((d >> 1) * 2);
  float ang = (float)t * __expf(-j2 * (9.210340371976184f / 128.0f));
  spa += (d & 1) ? __cosf(ang) : __sinf(ang);
  spa_seq[bt * Dm + d] = spa;
}

// ---------------- semantic embedding init ----------------

__global__ __launch_bounds__(128) void sem_init_kernel(
    const float* __restrict__ token_emb, const float* __restrict__ seg_emb,
    const float* __restrict__ pos_emb, const int* __restrict__ nodes,
    const int* __restrict__ cat_lut, const int* __restrict__ sub_lut,
    float* __restrict__ sem) {
  int bt = blockIdx.x;
  int d = threadIdx.x;
  int n = nodes[bt];
  int t = bt & (Tseq - 1);
  sem[bt * Dm + d] = token_emb[sub_lut[n] * Dm + d] +
                     seg_emb[cat_lut[n] * Dm + d] + pos_emb[t * Dm + d];
}

// ---------------- fused per-batch MHA (TDM stage + WMMA) ----------------
// one block == one batch element; 8 waves, each owns a 16-row query slab.
// dynamic LDS (288 KB):
//   [0,64K)     Xq  f32  (overlaid by S after QKV)
//   [64K,128K)  Xkv f32  (dead after QKV)
//   [128K,160K) Q f16 ; [160K,192K) K f16 ; [192K,224K) V f16
//   [0,64K)     S   f32  (per-head scores / probs)
//   [224K,288K) O   f32  (concat-head attention output)
__global__ __launch_bounds__(256) void mha_kernel(
    const float* __restrict__ qsrc, const float* __restrict__ kvsrc,
    const float* __restrict__ in_w, const float* __restrict__ in_b,
    const float* __restrict__ out_w, const float* __restrict__ out_b,
    const int* __restrict__ lengths, float* __restrict__ out) {
  extern __shared__ char smem[];
  float* Xq = (float*)(smem);
  float* Xkv = (float*)(smem + 65536);
  _Float16* Qs = (_Float16*)(smem + 131072);
  _Float16* Ks = (_Float16*)(smem + 163840);
  _Float16* Vs = (_Float16*)(smem + 196608);
  float* S = (float*)(smem);
  float* O = (float*)(smem + 229376);

  int b = blockIdx.x;
  int tid = threadIdx.x;
  int wave = tid >> 5;
  int lane = tid & 31;
  int row0 = wave * 16;

  // async DMA both activation tiles into LDS (contiguous 128x128 f32)
  if (wave == 0) {
    tdm_load_tile_f32(lds_offset_of(Xq), qsrc + (size_t)b * Tseq * Dm,
                      Tseq * Dm, 1, Tseq * Dm);
    tdm_load_tile_f32(lds_offset_of(Xkv), kvsrc + (size_t)b * Tseq * Dm,
                      Tseq * Dm, 1, Tseq * Dm);
    __builtin_amdgcn_s_wait_tensorcnt(0);
  }
  __syncthreads();

  // ---- QKV = [Xq|Xkv] @ in_w^T + in_b  -> f16 LDS ----
  v16h aq[4], akv[4];
#pragma unroll
  for (int kk = 0; kk < 4; ++kk) {
    aq[kk] = loadA_f32cvt(Xq, Dm, row0, kk * 32);
    akv[kk] = loadA_f32cvt(Xkv, Dm, row0, kk * 32);
  }
  for (int n0 = 0; n0 < 3 * Dm; n0 += 16) {
    v8f c = {};
    const v16h* asrc = (n0 < Dm) ? aq : akv;
#pragma unroll
    for (int kk = 0; kk < 4; ++kk) {
      v16h bf = loadB_wT(in_w, Dm, n0, kk * 32);
      c = wmma_f16(asrc[kk], bf, c);
    }
    _Float16* dst = (n0 < Dm) ? Qs : ((n0 < 2 * Dm) ? Ks : Vs);
    int col0 = n0 & (Dm - 1);
    int n = lane & 15, hf = lane >> 4;
#pragma unroll
    for (int v = 0; v < 8; ++v)
      dst[(row0 + v + hf * 8) * Dm + col0 + n] =
          (_Float16)(c[v] + in_b[n0 + n]);
  }
  __syncthreads();  // QKV ready; Xq/Xkv dead -> S may overlay

  int len = lengths[b];
  for (int h = 0; h < 8; ++h) {
    // scores: S = (Q_h @ K_h^T) / sqrt(16)
    v16h aqh = loadA_f16_pad(Qs, Dm, row0, h * 16, 16);
    for (int n0 = 0; n0 < Tseq; n0 += 16) {
      v8f c = {};
      v16h bk = loadB_nk_f16_pad(Ks, Dm, n0, h * 16, 16);
      c = wmma_f16(aqh, bk, c);
      storeC_f32(S, Tseq, row0, n0, c * 0.25f);
    }
    __syncthreads();
    // masked softmax, one thread per query row
    if (tid < Tseq) {
      float mx = -1e30f;
      for (int j = 0; j < len; ++j) mx = fmaxf(mx, S[tid * Tseq + j]);
      float sum = 0.0f;
      for (int j = 0; j < Tseq; ++j) {
        float e = (j < len) ? __expf(S[tid * Tseq + j] - mx) : 0.0f;
        S[tid * Tseq + j] = e;
        sum += e;
      }
      float inv = 1.0f / sum;
      for (int j = 0; j < Tseq; ++j) S[tid * Tseq + j] *= inv;
    }
    __syncthreads();
    // O_h = P @ V_h
    v8f c = {};
#pragma unroll
    for (int kk = 0; kk < 4; ++kk) {
      v16h ap = loadA_f32cvt(S, Tseq, row0, kk * 32);
      v16h bv = loadB_kn_f16(Vs, Dm, h * 16, kk * 32);
      c = wmma_f16(ap, bv, c);
    }
    storeC_f32(O, Dm, row0, h * 16, c);
    __syncthreads();  // all P reads done before next head's scores
  }

  // ---- out projection: Y = O @ out_w^T + out_b ----
  v16h ao[4];
#pragma unroll
  for (int kk = 0; kk < 4; ++kk) ao[kk] = loadA_f32cvt(O, Dm, row0, kk * 32);
  for (int n0 = 0; n0 < Dm; n0 += 16) {
    v8f c = {};
#pragma unroll
    for (int kk = 0; kk < 4; ++kk) {
      v16h bw = loadB_wT(out_w, Dm, n0, kk * 32);
      c = wmma_f16(ao[kk], bw, c);
    }
    int n = lane & 15, hf = lane >> 4;
#pragma unroll
    for (int v = 0; v < 8; ++v)
      out[(size_t)b * Tseq * Dm + (row0 + v + hf * 8) * Dm + n0 + n] =
          c[v] + out_b[n0 + n];
  }
}

// ---------------- generic WMMA GEMM: Y = act(X @ W^T + bias) ----------------
// grid (M/128, N/64), 256 threads; A (128x32) and B (64x32) tiles staged in
// LDS by the Tensor Data Mover each k-step.
__global__ __launch_bounds__(256) void gemm_kernel(
    const float* __restrict__ X, const float* __restrict__ W,
    const float* __restrict__ bias, float* __restrict__ Y, int M, int N,
    int Kd, int act) {
  __shared__ float Asf[128 * 32];  // 16 KB
  __shared__ float Bsf[64 * 32];   //  8 KB
  int rb = blockIdx.x * 128;
  int cb = blockIdx.y * 64;
  int tid = threadIdx.x;
  int wave = tid >> 5;
  int lane = tid & 31;
  v8f acc[4] = {{}, {}, {}, {}};
  for (int k0 = 0; k0 < Kd; k0 += 32) {
    if (wave == 0) {
      tdm_load_tile_f32(lds_offset_of(Asf), X + (size_t)rb * Kd + k0, 32, 128,
                        (unsigned)Kd);
      tdm_load_tile_f32(lds_offset_of(Bsf), W + (size_t)cb * Kd + k0, 32, 64,
                        (unsigned)Kd);
      __builtin_amdgcn_s_wait_tensorcnt(0);
    }
    __syncthreads();
    v16h a = loadA_f32cvt(Asf, 32, wave * 16, 0);
#pragma unroll
    for (int nt = 0; nt < 4; ++nt) {
      v16h bf = loadB_wT(Bsf, 32, nt * 16, 0);
      acc[nt] = wmma_f16(a, bf, acc[nt]);
    }
    __syncthreads();
  }
  int n = lane & 15, hf = lane >> 4;
#pragma unroll
  for (int nt = 0; nt < 4; ++nt)
#pragma unroll
    for (int v = 0; v < 8; ++v) {
      int col = cb + nt * 16 + n;
      float val = acc[nt][v] + bias[col];
      if (act == 1)
        val = fmaxf(val, 0.0f);
      else if (act == 2)
        val = gelu_tanh(val);
      Y[(size_t)(rb + wave * 16 + v + hf * 8) * N + col] = val;
    }
}

// ---------------- residual + LayerNorm (R may be null) ----------------

__global__ __launch_bounds__(128) void add_ln_kernel(
    const float* __restrict__ A, const float* __restrict__ R,
    const float* __restrict__ g, const float* __restrict__ be,
    float* __restrict__ Yo) {
  __shared__ float red[Dm];
  int row = blockIdx.x;
  int d = threadIdx.x;
  float x = A[row * Dm + d] + (R ? R[row * Dm + d] : 0.0f);
  float m = block_sum128(x, red) * (1.0f / Dm);
  float dv = x - m;
  float var = block_sum128(dv * dv, red) * (1.0f / Dm);
  Yo[row * Dm + d] = dv * rsqrtf(var + 1e-5f) * g[d] + be[d];
}

// ---------------- fusion concat with fixed weights ----------------

__global__ __launch_bounds__(256) void concat_scale_kernel(
    const float* __restrict__ s2s, const float* __restrict__ sp2,
    float* __restrict__ cat) {
  int bt = blockIdx.x;
  int i = threadIdx.x;
  float v = (i < Dm) ? W_SEM_C * s2s[bt * Dm + i]
                     : W_SPA_C * sp2[bt * Dm + (i - Dm)];
  cat[bt * 2 * Dm + i] = v;
}

// ---------------- masked mean pool ----------------

__global__ __launch_bounds__(128) void pool_kernel(
    const float* __restrict__ fused, const int* __restrict__ lengths,
    float* __restrict__ out) {
  int b = blockIdx.x;
  int d = threadIdx.x;
  int len = lengths[b];
  float s = 0.0f;
  for (int t = 0; t < len; ++t) s += fused[(b * Tseq + t) * Dm + d];
  out[b * Dm + d] = s / (float)(len > 0 ? len : 1);
}

// ---------------- host orchestration ----------------

extern "C" void kernel_launch(void* const* d_in, const int* in_sizes, int n_in,
                              void* d_out, int out_size, void* d_ws,
                              size_t ws_size, hipStream_t stream) {
  const float* node_emb = (const float*)d_in[0];
  const float* rel_emb = (const float*)d_in[1];
  const float* token_emb = (const float*)d_in[2];
  const float* segment_emb = (const float*)d_in[3];
  const float* pos_emb = (const float*)d_in[4];
  const float* enc_in_w = (const float*)d_in[5];    // [2,384,128]
  const float* enc_in_b = (const float*)d_in[6];    // [2,384]
  const float* enc_out_w = (const float*)d_in[7];   // [2,128,128]
  const float* enc_out_b = (const float*)d_in[8];
  const float* enc_l1_w = (const float*)d_in[9];    // [2,512,128]
  const float* enc_l1_b = (const float*)d_in[10];
  const float* enc_l2_w = (const float*)d_in[11];   // [2,128,512]
  const float* enc_l2_b = (const float*)d_in[12];
  const float* enc_ln1_g = (const float*)d_in[13];
  const float* enc_ln1_b = (const float*)d_in[14];
  const float* enc_ln2_g = (const float*)d_in[15];
  const float* enc_ln2_b = (const float*)d_in[16];
  const float* co_sem_in_w = (const float*)d_in[17];
  const float* co_sem_in_b = (const float*)d_in[18];
  const float* co_sem_out_w = (const float*)d_in[19];
  const float* co_sem_out_b = (const float*)d_in[20];
  const float* co_spa_in_w = (const float*)d_in[21];
  const float* co_spa_in_b = (const float*)d_in[22];
  const float* co_spa_out_w = (const float*)d_in[23];
  const float* co_spa_out_b = (const float*)d_in[24];
  const float* fuse_w = (const float*)d_in[25];     // [128,256]
  const float* fuse_b = (const float*)d_in[26];
  const float* fuse_ln_g = (const float*)d_in[27];
  const float* fuse_ln_b = (const float*)d_in[28];
  const float* ffn_w1 = (const float*)d_in[29];     // [512,128]
  const float* ffn_b1 = (const float*)d_in[30];
  const float* ffn_w2 = (const float*)d_in[31];     // [128,512]
  const float* ffn_b2 = (const float*)d_in[32];
  const float* ffn_ln_g = (const float*)d_in[33];
  const float* ffn_ln_b = (const float*)d_in[34];
  const int* nodes = (const int*)d_in[35];          // [32,128]
  const int* lengths = (const int*)d_in[36];        // [32]
  const int* nbr_idx = (const int*)d_in[37];        // [50000,10]
  const int* nbr_rel = (const int*)d_in[38];
  const int* cat_lut = (const int*)d_in[39];
  const int* sub_lut = (const int*)d_in[40];

  float* ws = (float*)d_ws;
  const int BT = Bsz * Tseq;  // 4096
  float* spa = ws;                       // 4096*128
  float* sem = ws + 524288;              // 4096*128
  float* t1 = ws + 1048576;              // 4096*128
  float* hid = ws + 1572864;             // 4096*512
  float* t3 = ws + 3670016;              // 4096*128
  float* catb = ws + 4194304;            // 4096*256
  float* fus = ws + 5242880;             // 4096*128
  float* t4 = ws + 5767168;              // 4096*128

  (void)hipFuncSetAttribute((const void*)mha_kernel,
                            hipFuncAttributeMaxDynamicSharedMemorySize,
                            MHA_LDS);

  spatial_kernel<<<BT, 128, 0, stream>>>(node_emb, rel_emb, nodes, nbr_idx,
                                         nbr_rel, spa);
  sem_init_kernel<<<BT, 128, 0, stream>>>(token_emb, segment_emb, pos_emb,
                                          nodes, cat_lut, sub_lut, sem);
  for (int l = 0; l < 2; ++l) {
    mha_kernel<<<Bsz, 256, MHA_LDS, stream>>>(
        sem, sem, enc_in_w + l * 384 * 128, enc_in_b + l * 384,
        enc_out_w + l * 128 * 128, enc_out_b + l * 128, lengths, t1);
    add_ln_kernel<<<BT, 128, 0, stream>>>(sem, t1, enc_ln1_g + l * 128,
                                          enc_ln1_b + l * 128, sem);
    gemm_kernel<<<dim3(32, 8), 256, 0, stream>>>(
        sem, enc_l1_w + l * 512 * 128, enc_l1_b + l * 512, hid, BT, 512, 128,
        1);
    gemm_kernel<<<dim3(32, 2), 256, 0, stream>>>(
        hid, enc_l2_w + l * 128 * 512, enc_l2_b + l * 128, t1, BT, 128, 512,
        0);
    add_ln_kernel<<<BT, 128, 0, stream>>>(sem, t1, enc_ln2_g + l * 128,
                                          enc_ln2_b + l * 128, sem);
  }
  // dual cross-attention
  mha_kernel<<<Bsz, 256, MHA_LDS, stream>>>(sem, spa, co_sem_in_w, co_sem_in_b,
                                            co_sem_out_w, co_sem_out_b,
                                            lengths, t1);  // sem2spa
  mha_kernel<<<Bsz, 256, MHA_LDS, stream>>>(spa, sem, co_spa_in_w, co_spa_in_b,
                                            co_spa_out_w, co_spa_out_b,
                                            lengths, t3);  // spa2sem
  concat_scale_kernel<<<BT, 256, 0, stream>>>(t1, t3, catb);
  gemm_kernel<<<dim3(32, 2), 256, 0, stream>>>(catb, fuse_w, fuse_b, t4, BT,
                                               128, 256, 0);
  add_ln_kernel<<<BT, 128, 0, stream>>>(t4, nullptr, fuse_ln_g, fuse_ln_b,
                                        fus);
  gemm_kernel<<<dim3(32, 8), 256, 0, stream>>>(fus, ffn_w1, ffn_b1, hid, BT,
                                               512, 128, 2);
  gemm_kernel<<<dim3(32, 2), 256, 0, stream>>>(hid, ffn_w2, ffn_b2, t1, BT,
                                               128, 512, 0);
  add_ln_kernel<<<BT, 128, 0, stream>>>(fus, t1, ffn_ln_g, ffn_ln_b, fus);
  pool_kernel<<<Bsz, 128, 0, stream>>>(fus, lengths, (float*)d_out);
}
